// EnhancedInformerAutoformerHybrid_43190191129174
// MI455X (gfx1250) — matrix operand order repaired and checked by reading, not compiled, not run on hardware
//
#include <hip/hip_runtime.h>
#include <hip/hip_bf16.h>
#include <stdint.h>

// Problem constants (match reference)
#define BB   8
#define LL   4096
#define DD   1024
#define HH   16
#define DKK  64
#define MAXF 10
#define MINF 3
#define MTOT (BB * LL)          // 32768 rows in the projection GEMMs

typedef __attribute__((ext_vector_type(16))) __bf16 v16bf;
typedef __attribute__((ext_vector_type(8)))  float  v8f;
typedef __attribute__((ext_vector_type(4)))  unsigned int v4u;
typedef __attribute__((ext_vector_type(8)))  int v8i;
typedef __attribute__((ext_vector_type(4)))  int v4i;

// ---------------- fp32 -> bf16 (RNE) ----------------
__global__ void cast_bf16_kernel(const float* __restrict__ src,
                                 uint16_t* __restrict__ dst, int n) {
  int i = blockIdx.x * blockDim.x + threadIdx.x;
  if (i >= n) return;
  uint32_t u = __float_as_uint(src[i]);
  uint32_t r = u + 0x7fffu + ((u >> 16) & 1u);
  if ((u & 0x7f800000u) == 0x7f800000u) r = u;  // keep inf/nan payload
  dst[i] = (uint16_t)(r >> 16);
}

// ---------------- Tensor Data Mover: 2-D bf16 tile -> LDS -----------------
// Builds D# group0 (128b) + group1 (256b) per CDNA5 ISA §8.3/8.4 and issues
// TENSOR_LOAD_TO_LDS.  Pad: every 32 DWORDs (one 128B row) insert 4 DWORDs
// (16B) -> LDS row stride 144B (bank-conflict-free fragment reads).
#define GEMM_BK        64
#define LDS_ROW_BYTES  144                    // 128 data + 16 pad
#define TILE_BYTES     (128 * LDS_ROW_BYTES)  // 18432 per tile buffer

__device__ __forceinline__ void tdm_load_tile_2d(const void* gptr,
                                                 uint32_t lds_off,
                                                 uint32_t tile_d0_elems,
                                                 uint32_t tile_d1_rows,
                                                 uint32_t row_stride_elems) {
  const uint64_t ga = (uint64_t)(uintptr_t)gptr;
  v4u g0;
  g0[0] = 1u;                                   // count=1, user descriptor
  g0[1] = lds_off;                              // LDS byte address
  g0[2] = (uint32_t)ga;                         // global_addr[31:0]
  g0[3] = (uint32_t)(ga >> 32) | (2u << 30);    // global_addr[56:32] | type=2

  const uint32_t td0 = 0x40000000u;             // huge tensor dims: tiles are
  const uint32_t td1 = 0x40000000u;             // always fully in-bounds
  const uint32_t pad_interval = 4u;             // 32 DWORDs between pads
  const uint32_t pad_amount   = 3u;             // 4 DWORDs of pad
  v8i g1;
  g1[0] = (int)((1u << 16) |                    // data_size = 2 bytes
                (1u << 20) |                    // pad_enable
                (pad_interval << 22) | (pad_amount << 25));
  g1[1] = (int)((td0 & 0xFFFFu) << 16);                       // dim0[15:0]
  g1[2] = (int)((td0 >> 16) | ((td1 & 0xFFFFu) << 16));       // dim0[31:16]|dim1[15:0]
  g1[3] = (int)((td1 >> 16) | (tile_d0_elems << 16));         // dim1[31:16]|tile_dim0
  g1[4] = (int)(tile_d1_rows);                                // tile_dim1 (tile_dim2=0)
  g1[5] = (int)row_stride_elems;                              // dim0_stride[31:0]
  g1[6] = 0;                                                  // stride bits: 0
  g1[7] = 0;
  const v4i z4 = {0, 0, 0, 0};
#if __clang_major__ >= 23
  const v8i z8 = {0, 0, 0, 0, 0, 0, 0, 0};
  __builtin_amdgcn_tensor_load_to_lds(g0, g1, z4, z4, z8, 0);
#else
  __builtin_amdgcn_tensor_load_to_lds(g0, g1, z4, z4, 0);
#endif
}

// ---------------- bf16 WMMA GEMM with TDM double-buffered LDS staging -----
// C[M,N] = A[M,K] * W[N,K]^T + bias.  M=32768, N=K=1024.
// Block = 128x128 C-tile, 256 threads (8 waves, 4(M) x 2(N)).
// Per k-step of 64: TDM loads A-tile(128x64) + W-tile(128x64) into LDS.
__global__ __launch_bounds__(256)
void gemm_bf16_kernel(const uint16_t* __restrict__ A,
                      const uint16_t* __restrict__ W,
                      const float* __restrict__ bias,
                      float* __restrict__ C) {
  extern __shared__ uint8_t smem[];   // [2 x A-tile][2 x W-tile] = 72 KB
  const uint32_t lds_base = (uint32_t)(uintptr_t)smem;  // addr[31:0] = LDS offset

  const int K = DD, N = DD;
  const int lane = threadIdx.x & 31;
  const int wave = threadIdx.x >> 5;
  const int wm   = wave & 3;     // 0..3
  const int wn   = wave >> 2;    // 0..1
  const int blockM = blockIdx.y * 128;
  const int blockN = blockIdx.x * 128;
  const int half = lane >> 4;    // K-half of the fragment
  const int l16  = lane & 15;    // M (A-frag) / N (B-frag) index

  v8f acc[2][4];
  #pragma unroll
  for (int i = 0; i < 2; i++)
    #pragma unroll
    for (int j = 0; j < 4; j++)
      #pragma unroll
      for (int e = 0; e < 8; e++) acc[i][j][e] = 0.0f;

  union Frag { v16bf v; uint4 u[2]; };

  const uint32_t offA[2] = {0u, (uint32_t)TILE_BYTES};
  const uint32_t offB[2] = {(uint32_t)(2 * TILE_BYTES), (uint32_t)(3 * TILE_BYTES)};

  // prologue: stage k-step 0 into buffer 0
  if (wave == 0) {
    tdm_load_tile_2d(A + (size_t)blockM * K, lds_base + offA[0], GEMM_BK, 128, K);
    tdm_load_tile_2d(W + (size_t)blockN * K, lds_base + offB[0], GEMM_BK, 128, K);
  }

  const int KSTEPS = K / GEMM_BK;   // 16
  for (int kt = 0; kt < KSTEPS; kt++) {
    const int cur = kt & 1;
    __builtin_amdgcn_s_wait_tensorcnt((short)0);  // no-op for non-issuing waves
    __syncthreads();                               // tile[cur] ready; tile[!cur] free
    if (wave == 0 && kt + 1 < KSTEPS) {
      const int k0 = (kt + 1) * GEMM_BK;
      tdm_load_tile_2d(A + (size_t)blockM * K + k0, lds_base + offA[1 - cur],
                       GEMM_BK, 128, K);
      tdm_load_tile_2d(W + (size_t)blockN * K + k0, lds_base + offB[1 - cur],
                       GEMM_BK, 128, K);
    }

    const uint8_t* aBuf = smem + offA[cur];
    const uint8_t* bBuf = smem + offB[cur];
    #pragma unroll
    for (int ks = 0; ks < 2; ks++) {              // two 32-wide K slices
      Frag a[2], b[4];
      // A-frag: elements 0..7 = K[ks*32+half*8 ..], 8..15 = +16
      #pragma unroll
      for (int i = 0; i < 2; i++) {
        const uint8_t* row = aBuf + (size_t)(wm * 32 + i * 16 + l16) * LDS_ROW_BYTES;
        a[i].u[0] = *reinterpret_cast<const uint4*>(row + (ks * 32 + half * 8) * 2);
        a[i].u[1] = *reinterpret_cast<const uint4*>(row + (ks * 32 + half * 8) * 2 + 32);
      }
      // B-frag: elements 0..7 = K[ks*32+half*8 ..], 8..15 = +16
      #pragma unroll
      for (int j = 0; j < 4; j++) {
        const uint8_t* row = bBuf + (size_t)(wn * 64 + j * 16 + l16) * LDS_ROW_BYTES;
        b[j].u[0] = *reinterpret_cast<const uint4*>(row + (ks * 32 + half * 8) * 2);
        b[j].u[1] = *reinterpret_cast<const uint4*>(row + (ks * 32 + half * 8) * 2 + 32);
      }
      #pragma unroll
      for (int i = 0; i < 2; i++)
        #pragma unroll
        for (int j = 0; j < 4; j++)
          acc[i][j] = __builtin_amdgcn_wmma_f32_16x16x32_bf16(
              false, a[i].v, false, b[j].v, (short)0, acc[i][j], false, false);
    }
  }

  // C/D layout: VGPR r -> row (r + 8*half), lanes hold 16 consecutive columns
  #pragma unroll
  for (int i = 0; i < 2; i++) {
    #pragma unroll
    for (int j = 0; j < 4; j++) {
      const int col = blockN + wn * 64 + j * 16 + l16;
      const float bv = bias[col];
      #pragma unroll
      for (int r = 0; r < 8; r++) {
        const int row = blockM + wm * 32 + i * 16 + r + half * 8;
        C[(size_t)row * N + col] = acc[i][j][r] + bv;
      }
    }
  }
}

// ---------------- wave reductions (wave32) ----------------
__device__ __forceinline__ float wred_sum(float v) {
  for (int m = 16; m; m >>= 1) v += __shfl_xor(v, m, 32);
  return v;
}
__device__ __forceinline__ float wred_max(float v) {
  for (int m = 16; m; m >>= 1) v = fmaxf(v, __shfl_xor(v, m, 32));
  return v;
}

// ---------------- q_scores: one wave per (b,h,l) row of 64 ----------------
__global__ __launch_bounds__(256)
void qscore_kernel(const float* __restrict__ Q, float* __restrict__ qs) {
  const int wave = threadIdx.x >> 5;
  const int lane = threadIdx.x & 31;
  const int r = blockIdx.x * 8 + wave;        // 0 .. B*H*L-1
  const int b = r / (HH * LL);
  const int rem = r % (HH * LL);
  const int h = rem / LL;
  const int l = rem % LL;
  const float* q = Q + ((size_t)b * LL + l) * DD + h * DKK;
  const float q0 = q[lane], q1 = q[lane + 32];

  float s1 = wred_sum(q0 + q1);
  float s2 = wred_sum(q0 * q0 + q1 * q1);
  float mx = wred_max(fmaxf(q0, q1));
  float se = wred_sum(__expf(q0 - mx) + __expf(q1 - mx));
  const float lse = mx + __logf(se);
  const float p0 = __expf(q0 - lse), p1 = __expf(q1 - lse);
  float ent = wred_sum(-(p0 * __logf(p0 + 1e-9f) + p1 * __logf(p1 + 1e-9f)));

  if (lane == 0) {
    const float mean = s1 * (1.0f / DKK);
    const float var  = (s2 - DKK * mean * mean) * (1.0f / (DKK - 1));
    const float l2   = sqrtf(s2);
    qs[(size_t)r] = 0.5f * l2 + 0.3f * ent + 0.2f * var;
  }
}

// ---------------- per-(b,h) mean/std -> factor; u[b] = factor[b,0] --------
__global__ __launch_bounds__(256)
void stats_kernel(const float* __restrict__ qs, int* __restrict__ factor,
                  int* __restrict__ u) {
  const int bh = blockIdx.x;                  // 0..127
  const float* s = qs + (size_t)bh * LL;
  float s1 = 0.f, s2 = 0.f;
  for (int i = threadIdx.x; i < LL; i += 256) { float v = s[i]; s1 += v; s2 += v * v; }
  __shared__ float sh1[256], sh2[256];
  sh1[threadIdx.x] = s1; sh2[threadIdx.x] = s2; __syncthreads();
  for (int o = 128; o > 0; o >>= 1) {
    if (threadIdx.x < o) { sh1[threadIdx.x] += sh1[threadIdx.x + o];
                           sh2[threadIdx.x] += sh2[threadIdx.x + o]; }
    __syncthreads();
  }
  if (threadIdx.x == 0) {
    const float mean = sh1[0] * (1.0f / LL);
    const float var  = (sh2[0] - (float)LL * mean * mean) * (1.0f / (LL - 1));
    const float sd   = sqrtf(fmaxf(var, 0.f));
    float f = rintf(sd / (mean + 1e-6f) * (float)MAXF);
    f = fminf(fmaxf(f, (float)MINF), (float)MAXF);
    const int fi = (int)f;
    factor[bh] = fi;
    if ((bh % HH) == 0) u[bh / HH] = fi;
  }
}

// ---------------- top-10 per (b,h): one wave, 10 argmax passes ------------
__global__ __launch_bounds__(32)
void topk_kernel(const float* __restrict__ qs, int* __restrict__ topidx) {
  const int bh = blockIdx.x;
  const float* s = qs + (size_t)bh * LL;
  const int lane = threadIdx.x;
  __shared__ int chosen[MAXF];
  for (int t = 0; t < MAXF; t++) {
    float bv = -3.4e38f; int bi = LL;
    for (int l = lane; l < LL; l += 32) {
      bool skip = false;
      for (int c = 0; c < t; c++) skip |= (chosen[c] == l);
      if (skip) continue;
      const float v = s[l];
      if (v > bv || (v == bv && l < bi)) { bv = v; bi = l; }
    }
    for (int m = 16; m; m >>= 1) {
      const float ov = __shfl_xor(bv, m, 32);
      const int   oi = __shfl_xor(bi, m, 32);
      if (ov > bv || (ov == bv && oi < bi)) { bv = ov; bi = oi; }
    }
    if (lane == 0) { chosen[t] = bi; topidx[bh * MAXF + t] = bi; }
    __syncthreads();
  }
}

// ---------------- sparse attention: block = (b,h, pair of query rows) -----
__global__ __launch_bounds__(256)
void attn_kernel(const float* __restrict__ Q, const float* __restrict__ Km,
                 const float* __restrict__ Vm, const int* __restrict__ topidx,
                 float* __restrict__ oattn) {
  __shared__ float sc[2][LL];       // scores for 2 query rows (32 KB)
  __shared__ float qsm[2][DKK];
  const int bh = blockIdx.x;
  const int pg = blockIdx.y;        // 0..4 -> rows 2*pg, 2*pg+1
  const int b = bh / HH, h = bh % HH;
  const int tid = threadIdx.x;

  for (int i = tid; i < 2 * DKK; i += 256) {
    const int p = i / DKK, k = i % DKK;
    const int l = topidx[bh * MAXF + 2 * pg + p];
    qsm[p][k] = Q[((size_t)b * LL + l) * DD + h * DKK + k];
  }
  __syncthreads();

  const float scale = 0.125f;       // 1/sqrt(64)
  for (int l = tid; l < LL; l += 256) {
    const float* kr = Km + ((size_t)b * LL + l) * DD + h * DKK;
    float a0 = 0.f, a1 = 0.f;
    #pragma unroll
    for (int k = 0; k < DKK; k += 4) {
      const float4 kv = *reinterpret_cast<const float4*>(kr + k);
      a0 += kv.x * qsm[0][k] + kv.y * qsm[0][k + 1] + kv.z * qsm[0][k + 2] + kv.w * qsm[0][k + 3];
      a1 += kv.x * qsm[1][k] + kv.y * qsm[1][k + 1] + kv.z * qsm[1][k + 2] + kv.w * qsm[1][k + 3];
    }
    sc[0][l] = a0 * scale;
    sc[1][l] = a1 * scale;
  }
  __syncthreads();

  const int wave = tid >> 5, lane = tid & 31;
  if (wave < 2) {                    // softmax over L, one wave per row
    float mx = -3.4e38f;
    for (int l = lane; l < LL; l += 32) mx = fmaxf(mx, sc[wave][l]);
    mx = wred_max(mx);
    float sum = 0.f;
    for (int l = lane; l < LL; l += 32) {
      const float e = __expf(sc[wave][l] - mx);
      sc[wave][l] = e; sum += e;
    }
    sum = wred_sum(sum);
    const float inv = 1.0f / sum;
    for (int l = lane; l < LL; l += 32) sc[wave][l] *= inv;
  }
  __syncthreads();

  if (tid < 128) {                   // attn @ V : d = tid%64, p = tid/64
    const int d = tid & 63;
    const int p = tid >> 6;
    float acc = 0.f;
    for (int l = 0; l < LL; l++)
      acc += sc[p][l] * Vm[((size_t)b * LL + l) * DD + h * DKK + d];
    oattn[((size_t)bh * MAXF + 2 * pg + p) * DKK + d] = acc;
  }
}

// ---------------- output = bo (broadcast) ----------------
__global__ void fill_bias_kernel(const float* __restrict__ bo,
                                 float* __restrict__ out, int n) {
  const int i = blockIdx.x * blockDim.x + threadIdx.x;
  if (i >= n) return;
  out[i] = bo[i & (DD - 1)];
}

// ---------------- sparse rank-64 updates: out[b,l,:] += oattn @ Wo_chunk^T
__global__ __launch_bounds__(256)
void scatter_kernel(const float* __restrict__ oattn, const int* __restrict__ topidx,
                    const int* __restrict__ u, const float* __restrict__ Wo,
                    float* __restrict__ out) {
  const int idx = blockIdx.x;           // 0 .. B*H*MAXF-1
  const int p  = idx % MAXF;
  const int bh = idx / MAXF;
  const int b = bh / HH, h = bh % HH;
  if (p >= u[b]) return;
  const int l = topidx[bh * MAXF + p];
  __shared__ float oa[DKK];
  if (threadIdx.x < DKK)
    oa[threadIdx.x] = oattn[((size_t)bh * MAXF + p) * DKK + threadIdx.x];
  __syncthreads();
  for (int d = threadIdx.x; d < DD; d += 256) {
    const float* w = Wo + (size_t)d * DD + h * DKK;
    float c = 0.f;
    #pragma unroll
    for (int k = 0; k < DKK; k++) c += oa[k] * w[k];
    atomicAdd(&out[((size_t)b * LL + l) * DD + d], c);
  }
}

// ---------------- host orchestration ----------------
extern "C" void kernel_launch(void* const* d_in, const int* in_sizes, int n_in,
                              void* d_out, int out_size, void* d_ws, size_t ws_size,
                              hipStream_t stream) {
  const float* x  = (const float*)d_in[0];
  const float* Wq = (const float*)d_in[1];
  const float* bq = (const float*)d_in[2];
  const float* Wk = (const float*)d_in[3];
  const float* bk = (const float*)d_in[4];
  const float* Wv = (const float*)d_in[5];
  const float* bv = (const float*)d_in[6];
  const float* Wo = (const float*)d_in[7];
  const float* bo = (const float*)d_in[8];
  float* out = (float*)d_out;

  char* ws = (char*)d_ws;
  size_t off = 0;
  auto alloc = [&](size_t bytes) -> char* {
    char* p = ws + off;
    off = (off + bytes + 255) & ~(size_t)255;
    return p;
  };

  const int nX = BB * LL * DD;          // 33,554,432
  const int nW = DD * DD;               // 1,048,576

  uint16_t* xb  = (uint16_t*)alloc((size_t)nX * 2);
  uint16_t* wqb = (uint16_t*)alloc((size_t)nW * 2);
  uint16_t* wkb = (uint16_t*)alloc((size_t)nW * 2);
  uint16_t* wvb = (uint16_t*)alloc((size_t)nW * 2);
  float* Qf = (float*)alloc((size_t)nX * 4);
  float* Kf = (float*)alloc((size_t)nX * 4);
  float* Vf = (float*)alloc((size_t)nX * 4);
  float* qs = (float*)alloc((size_t)BB * HH * LL * 4);
  int* topidx = (int*)alloc((size_t)BB * HH * MAXF * 4);
  int* factor = (int*)alloc((size_t)BB * HH * 4);
  int* uu     = (int*)alloc((size_t)BB * 4);
  float* oattn = (float*)alloc((size_t)BB * HH * MAXF * DKK * 4);
  (void)ws_size; (void)in_sizes; (void)n_in; (void)out_size;

  // 1) downconvert activations + projection weights to bf16
  cast_bf16_kernel<<<nX / 256, 256, 0, stream>>>(x,  xb,  nX);
  cast_bf16_kernel<<<nW / 256, 256, 0, stream>>>(Wq, wqb, nW);
  cast_bf16_kernel<<<nW / 256, 256, 0, stream>>>(Wk, wkb, nW);
  cast_bf16_kernel<<<nW / 256, 256, 0, stream>>>(Wv, wvb, nW);

  // 2) Q/K/V projections: TDM-staged double-buffered WMMA bf16 GEMM
  const dim3 ggrid(DD / 128, MTOT / 128);   // (8, 256)
  const size_t gemm_lds = 4 * TILE_BYTES;   // 73728 B
  gemm_bf16_kernel<<<ggrid, 256, gemm_lds, stream>>>(xb, wqb, bq, Qf);
  gemm_bf16_kernel<<<ggrid, 256, gemm_lds, stream>>>(xb, wkb, bk, Kf);
  gemm_bf16_kernel<<<ggrid, 256, gemm_lds, stream>>>(xb, wvb, bv, Vf);

  // 3) q_scores, 4) factor/u, 5) top-10
  qscore_kernel<<<(BB * HH * LL) / 8, 256, 0, stream>>>(Qf, qs);
  stats_kernel<<<BB * HH, 256, 0, stream>>>(qs, factor, uu);
  topk_kernel<<<BB * HH, 32, 0, stream>>>(qs, topidx);

  // 6) sparse attention (10 query rows per head, 2 per block)
  attn_kernel<<<dim3(BB * HH, MAXF / 2), 256, 0, stream>>>(Qf, Kf, Vf, topidx, oattn);

  // 7) output projection exploiting scatter sparsity
  fill_bias_kernel<<<nX / 256, 256, 0, stream>>>(bo, out, nX);
  scatter_kernel<<<BB * HH * MAXF, 256, 0, stream>>>(oattn, topidx, uu, Wo, out);
}
